// TopoLSTM_13426067768013
// MI455X (gfx1250) — matrix operand order repaired and checked by reading, not compiled, not run on hardware
//
#include <hip/hip_runtime.h>

#define TT 200
#define NN 256
#define HH 128
#define G4 512          // 4*H
#define VV 50000

typedef __attribute__((ext_vector_type(16))) __bf16 v16bf;
typedef __attribute__((ext_vector_type(2)))  __bf16 v2bf;
typedef __attribute__((ext_vector_type(8)))  float  v8f;

__device__ __forceinline__ v8f wmma_bf16(v16bf a, v16bf b, v8f c) {
  // D = A(16x32 bf16) * B(32x16 bf16) + C(16x16 f32)
  return __builtin_amdgcn_wmma_f32_16x16x32_bf16(false, a, false, b, (short)0, c,
                                                 false, false);
}

// Inverse of the 16-bit A-fragment layout (contiguous k order, matches B): within a 16x32 tile,
// element (m, kr) lives at lane = m + 16*hi, vector element e.
__device__ __forceinline__ void a_pos(int kr, int& hi, int& e) {
  hi = (kr >> 4) & 1;                 // kr in [16,32) -> upper half-wave
  e  = kr & 15;                       // kr%16
}

// ---------- B-matrix swizzle+convert: row-major f32 [128][pitch] -> fragment-major bf16 ----------
// dst[((ct*4 + kt)*32 + lane)*16 + e] = src[(kt*32 + e + (lane&16?16:0)) * pitch + ct*16 + (lane&15)]
__global__ void swz_b_kernel(const float* __restrict__ src, __bf16* __restrict__ dst,
                             int pitch) {
  int i = blockIdx.x * blockDim.x + threadIdx.x;
  const int total  = (pitch >> 4) * 4 * 512;
  const int stride = gridDim.x * blockDim.x;
  for (; i < total; i += stride) {
    const int e    = i & 15;
    const int lane = (i >> 4) & 31;
    const int kt   = (i >> 9) & 3;
    const int ct   = i >> 11;
    const int k    = kt * 32 + e + ((lane & 16) ? 16 : 0);
    const int n    = ct * 16 + (lane & 15);
    dst[i] = (__bf16)src[(size_t)k * pitch + n];
  }
}

// ---------------- gates_pre = gather(emb)[row] @ W + b ----------------
// rows = T*N = 51200 (row = t*N + n), block rt = 16 rows x 512 cols.
// Output stored in C/D-fragment order: gpSwz[((rt*32 + ct)*32 + lane)*8 + r]
__global__ __launch_bounds__(256) void gates_pre_kernel(
    const int* __restrict__ seq, const float* __restrict__ embTab,
    const __bf16* __restrict__ WbSwz, const float* __restrict__ b,
    float* __restrict__ gpSwz) {
  __shared__ __bf16 sA[4 * 32 * 16];   // A fragments, 16 rows x 128 K (4 k-tiles)
  const int tid  = threadIdx.x;
  const int lane = tid & 31;
  const int wave = tid >> 5;
  const size_t r0 = (size_t)blockIdx.x * 16;

  // gather 16 embedding rows straight into A-fragment-swizzled bf16 LDS
#pragma unroll
  for (int i = 0; i < 8; ++i) {
    const int idx = tid + i * 256;     // 0..2047
    const int rl  = idx >> 7;          // row (M) 0..15
    const int h   = idx & 127;         // K
    const int v   = seq[r0 + rl];
    const float val = embTab[(size_t)v * HH + h];
    int hi, e; a_pos(h & 31, hi, e);
    sA[(((h >> 5) * 32) + rl + 16 * hi) * 16 + e] = (__bf16)val;
  }
  __syncthreads();

#pragma unroll
  for (int ct4 = 0; ct4 < 4; ++ct4) {
    const int tile = wave * 4 + ct4;   // 0..31
    const int col  = tile * 16 + (lane & 15);
    v8f c;
    const float bc = b[col];
#pragma unroll
    for (int r = 0; r < 8; ++r) c[r] = bc;
#pragma unroll
    for (int kk = 0; kk < 4; ++kk) {
      const v16bf a  = *(const v16bf*)&sA[(kk * 32 + lane) * 16];
      const v16bf bb = *(const v16bf*)&WbSwz[(((size_t)tile * 4 + kk) * 32 + lane) * 16];
      c = wmma_bf16(a, bb, c);
    }
    *(v8f*)&gpSwz[((((size_t)blockIdx.x) * 32 + tile) * 32 + lane) * 8] = c;
  }
}

// ---------------- persistent recurrence: one block = 16 sequences, 200 steps ----------------
__global__ __launch_bounds__(1024) void recur_kernel(
    const float* __restrict__ mask, const float* __restrict__ topo,
    const float* __restrict__ U, const float* __restrict__ gpSwz,
    v2bf* __restrict__ hcStore, __bf16* __restrict__ hMeanSwz) {
  __shared__ __bf16 sU[32 * 4 * 32 * 16];   // U in B-fragment order, 128 KB resident
  __shared__ __bf16 sHswz[4 * 32 * 16];     // h_sum as A fragments (bf16)
  __shared__ float  sG[32 * 32 * 8];        // gates as C/D fragments (32 KB)

  const int tid  = threadIdx.x;
  const int lane = tid & 31;
  const int wave = tid >> 5;                // 0..31
  const int nb   = blockIdx.x;              // 0..15
  const int n0   = nb * 16;

  // stage U (f32 row-major [128][512]) -> B-fragment swizzled bf16 LDS
  for (int i = tid; i < HH * G4; i += 1024) {
    const int k = i >> 9;                   // 0..127
    const int n = i & 511;
    const int lb = (n & 15) + ((k & 16) ? 16 : 0);
    const int e  = k & 15;
    sU[((((n >> 4) * 4) + (k >> 5)) * 32 + lb) * 16 + e] = (__bf16)U[i];
  }

  float hacc[2] = {0.f, 0.f};
  float csum[2];
  __syncthreads();

  for (int t = 0; t < TT; ++t) {
    // ---- phase 1: weighted aggregation over prior states (t' < t) ----
#pragma unroll
    for (int i = 0; i < 2; ++i) {
      const int idx = tid + i * 1024;       // 0..2047
      const int nl  = idx >> 7;             // local n 0..15
      const int h   = idx & 127;
      const int n   = n0 + nl;
      const float* tp  = topo + ((size_t)t * NN + n) * TT;
      const v2bf*  hcp = hcStore + (size_t)n * HH + h;
      float ah0 = 0.f, ah1 = 0.f, ac0 = 0.f, ac1 = 0.f;
      int q = 0;
      for (; q + 1 < t; q += 2) {
        const float w0 = tp[q], w1 = tp[q + 1];
        const v2bf p0 = hcp[(size_t)q * (NN * HH)];
        const v2bf p1 = hcp[(size_t)(q + 1) * (NN * HH)];
        ah0 += w0 * (float)p0[0]; ac0 += w0 * (float)p0[1];
        ah1 += w1 * (float)p1[0]; ac1 += w1 * (float)p1[1];
      }
      if (q < t) {
        const float w = tp[q];
        const v2bf p = hcp[(size_t)q * (NN * HH)];
        ah0 += w * (float)p[0]; ac0 += w * (float)p[1];
      }
      csum[i] = ac0 + ac1;                  // stays in-register for phase 3
      const float ah = ah0 + ah1;
      int hi, e; a_pos(h & 31, hi, e);
      sHswz[(((h >> 5) * 32) + nl + 16 * hi) * 16 + e] = (__bf16)ah;
    }
    __syncthreads();

    // ---- phase 2: gates = bf16(h_sum) @ U + gates_pre[t]  (16x512, K=128) ----
    {
      v8f c = *(const v8f*)&gpSwz[((((size_t)t * 16 + nb) * 32 + wave) * 32 + lane) * 8];
#pragma unroll
      for (int kk = 0; kk < 4; ++kk) {
        const v16bf a  = *(const v16bf*)&sHswz[(kk * 32 + lane) * 16];
        const v16bf bb = *(const v16bf*)&sU[((wave * 4 + kk) * 32 + lane) * 16];
        c = wmma_bf16(a, bb, c);
      }
      *(v8f*)&sG[(wave * 32 + lane) * 8] = c;
    }
    __syncthreads();

    // ---- phase 3: elementwise LSTM cell, packed bf16x2 state store ----
#pragma unroll
    for (int i = 0; i < 2; ++i) {
      const int idx = tid + i * 1024;
      const int nl  = idx >> 7;
      const int h   = idx & 127;
      const int n   = n0 + nl;
      const int lc  = (h & 15) + ((nl & 8) << 1);   // C-fragment lane
      const int rc  = nl & 7;                       // C-fragment register
      const int ctb = h >> 4;                       // col tile base within a gate
      const float gi = sG[((ctb +  0) * 32 + lc) * 8 + rc];
      const float gf = sG[((ctb +  8) * 32 + lc) * 8 + rc];
      const float go = sG[((ctb + 16) * 32 + lc) * 8 + rc];
      const float gg = sG[((ctb + 24) * 32 + lc) * 8 + rc];
      const float si = 1.f / (1.f + __expf(-gi));
      const float sf = 1.f / (1.f + __expf(-gf));
      const float so = 1.f / (1.f + __expf(-go));
      const float m  = mask[(size_t)t * NN + n];
      const float cv = m * (sf * csum[i] + si * tanhf(gg));
      const float hv = m * (so * tanhf(cv));
      v2bf pk; pk[0] = (__bf16)hv; pk[1] = (__bf16)cv;
      hcStore[((size_t)t * NN + n) * HH + h] = pk;
      hacc[i] += m * hv;
      // prefetch next step's topo row (gfx1250 global_prefetch_b8)
      if (t + 1 < TT && h < 8)
        __builtin_prefetch(topo + ((size_t)(t + 1) * NN + n) * TT + h * 25, 0, 0);
    }
    __syncthreads();
  }

  // ---- mean over time, written directly in A-fragment order ----
#pragma unroll
  for (int i = 0; i < 2; ++i) {
    const int idx = tid + i * 1024;
    const int nl  = idx >> 7;
    const int h   = idx & 127;
    const int n   = n0 + nl;
    float len = 0.f;
    for (int t = 0; t < TT; ++t) len += mask[(size_t)t * NN + n];
    int hi, e; a_pos(h & 31, hi, e);
    hMeanSwz[((((size_t)nb * 4) + (h >> 5)) * 32 + nl + 16 * hi) * 16 + e] =
        (__bf16)(hacc[i] / len);
  }
}

// ---------------- out = h_mean(256x128) @ W_out(128x50000) + b_out ----------------
// One wave owns one 16-col tile; B fragments loaded once, reused for all 16 M-tiles.
__global__ __launch_bounds__(256) void out_gemm_kernel(
    const __bf16* __restrict__ hmSwz, const __bf16* __restrict__ WobSwz,
    const float* __restrict__ bout, float* __restrict__ out) {
  const int tid  = threadIdx.x;
  const int lane = tid & 31;
  const int wave = tid >> 5;
  const int ct   = blockIdx.x * 8 + wave;   // wave-uniform guard -> EXEC stays full
  if (ct >= VV / 16) return;
  const int col = ct * 16 + (lane & 15);

  v16bf bf[4];
#pragma unroll
  for (int kk = 0; kk < 4; ++kk)
    bf[kk] = *(const v16bf*)&WobSwz[(((size_t)ct * 4 + kk) * 32 + lane) * 16];

  const float bb  = bout[col];
  const int rbase = (lane & 16) ? 8 : 0;

  for (int mt = 0; mt < 16; ++mt) {         // all 256 rows
    v8f c;
#pragma unroll
    for (int r = 0; r < 8; ++r) c[r] = bb;
#pragma unroll
    for (int kk = 0; kk < 4; ++kk) {
      const v16bf a = *(const v16bf*)&hmSwz[(((size_t)mt * 4 + kk) * 32 + lane) * 16];
      c = wmma_bf16(a, bf[kk], c);
    }
#pragma unroll
    for (int r = 0; r < 8; ++r)
      out[(size_t)(mt * 16 + rbase + r) * VV + col] = c[r];
  }
}

extern "C" void kernel_launch(void* const* d_in, const int* in_sizes, int n_in,
                              void* d_out, int out_size, void* d_ws, size_t ws_size,
                              hipStream_t stream) {
  (void)in_sizes; (void)n_in; (void)out_size; (void)ws_size;
  const int*   seq  = (const int*)  d_in[0];
  const float* mask = (const float*)d_in[1];
  const float* topo = (const float*)d_in[2];
  const float* W    = (const float*)d_in[3];
  const float* U    = (const float*)d_in[4];
  const float* b    = (const float*)d_in[5];
  const float* emb  = (const float*)d_in[6];
  const float* Wout = (const float*)d_in[7];
  const float* bout = (const float*)d_in[8];
  float* out = (float*)d_out;

  char* ws = (char*)d_ws;
  __bf16* WobSwz   = (__bf16*)(ws);               // 128*50000*2  = 12,800,000
  float*  gpSwz    = (float*) (ws + 12800000);    // 200*256*512*4 = 104,857,600
  v2bf*   hcStore  = (v2bf*)  (ws + 117657600);   // 200*256*128*4 = 26,214,400
  __bf16* hMeanSwz = (__bf16*)(ws + 143872000);   // 256*128*2 = 65,536
  __bf16* WbSwz    = (__bf16*)(ws + 143937536);   // 128*512*2 = 131,072

  swz_b_kernel<<<2048, 256, 0, stream>>>(Wout, WobSwz, VV);
  swz_b_kernel<<<64,   256, 0, stream>>>(W,    WbSwz,  G4);
  gates_pre_kernel<<<(TT * NN) / 16, 256, 0, stream>>>(seq, emb, WbSwz, b, gpSwz);
  recur_kernel<<<NN / 16, 1024, 0, stream>>>(mask, topo, U, gpSwz, hcStore, hMeanSwz);
  out_gemm_kernel<<<(VV / 16 + 7) / 8, 256, 0, stream>>>(hMeanSwz, WobSwz, bout, out);
}
